// FastHierarchicalHeteroGraphConv_64338610094389
// MI455X (gfx1250) — matrix operand order repaired and checked by reading, not compiled, not run on hardware
//
#include <hip/hip_runtime.h>

// ---------------------------------------------------------------------------
// CDNA5 (gfx1250) hetero-GraphSAGE forward.
//   GEMMs:   v_wmma_f32_16x16x32_bf16, wave32, 16x128 strip per wave,
//            templatized on dual-A to keep the inner loop branch-free.
//   Scatter: global_atomic_add_f32, 32 lanes/edge.
//   Weights: pre-swizzled to per-lane B-fragment layout (32B/lane loads),
//            L2-resident (21 x 32KB << 192MB L2).
// ---------------------------------------------------------------------------

typedef __bf16 v16bf __attribute__((ext_vector_type(16)));
typedef __bf16 v8bf  __attribute__((ext_vector_type(8)));
typedef float  v8f   __attribute__((ext_vector_type(8)));

__device__ __forceinline__ unsigned short f2bf(float x) {
    unsigned int u = __float_as_uint(x);
    unsigned int r = u + 0x7FFFu + ((u >> 16) & 1u);   // round-nearest-even
    return (unsigned short)(r >> 16);
}
__device__ __forceinline__ float bf2f(unsigned short h) {
    return __uint_as_float(((unsigned int)h) << 16);
}

// --------------------------- elementwise helpers ---------------------------

__global__ void zero_kernel(float* __restrict__ p, long n) {
    long i = (long)blockIdx.x * blockDim.x + threadIdx.x;
    if (i < n) p[i] = 0.0f;
}

__global__ void cvt_bf16_kernel(const float* __restrict__ in,
                                unsigned short* __restrict__ out, long n) {
    long i = (long)blockIdx.x * blockDim.x + threadIdx.x;
    if (i < n) out[i] = f2bf(in[i]);
}

// Swizzle a 128x128 row-major [K][N] f32 weight into bf16 B-fragments:
// out[((ct*4+ks)*32 + lane)*16 + i] = W[(ks*32 + (lane>=16?8:0) + (i&7) + (i>=8?16:0))*128 + ct*16 + (lane&15)]
__global__ void swz_kernel(const float* __restrict__ W,
                           unsigned short* __restrict__ out) {
    int tid = blockIdx.x * blockDim.x + threadIdx.x;  // 0..16383
    if (tid >= 16384) return;
    int i    = tid & 15;
    int lane = (tid >> 4) & 31;
    int ks   = (tid >> 9) & 3;
    int ct   = (tid >> 11) & 7;
    int k  = ks * 32 + ((lane >> 4) << 3) + (i & 7) + ((i >> 3) << 4);
    int nn = ct * 16 + (lane & 15);
    out[tid] = f2bf(W[k * 128 + nn]);
}

// --------------------------- graph aggregation -----------------------------

__global__ void deg_kernel(const int* __restrict__ dst, float* __restrict__ deg, int e) {
    int i = blockIdx.x * blockDim.x + threadIdx.x;
    if (i < e) atomicAdd(&deg[dst[i]], 1.0f);
}

// 32 lanes per edge; each lane handles 4 contiguous channels (8B bf16 load).
__global__ void scatter_kernel(const unsigned short* __restrict__ feat,
                               const int* __restrict__ src,
                               const int* __restrict__ dst,
                               float* __restrict__ agg, int e) {
    long tid  = (long)blockIdx.x * blockDim.x + threadIdx.x;
    long edge = tid >> 5;
    if (edge >= e) return;
    int lane = (int)(tid & 31);
    int s = src[edge], d = dst[edge];
    const unsigned short* sp = feat + (long)s * 128 + lane * 4;
    float*                dp = agg  + (long)d * 128 + lane * 4;
    unsigned long long pk = *(const unsigned long long*)sp;
#pragma unroll
    for (int i = 0; i < 4; i++) {
        unsigned short h = (unsigned short)((pk >> (16 * i)) & 0xFFFFu);
        atomicAdd(dp + i, bf2f(h));
    }
}

// agg / max(deg,1) -> bf16
__global__ void finalize_kernel(const float* __restrict__ agg,
                                const float* __restrict__ deg,
                                unsigned short* __restrict__ out, long n) {
    long i = (long)blockIdx.x * blockDim.x + threadIdx.x;
    if (i >= n) return;
    float d = deg[i >> 7];
    out[i] = f2bf(agg[i] / fmaxf(d, 1.0f));
}

// ReLU + LayerNorm over 128 channels; one wave32 per row, 4 channels/lane.
__global__ void relu_ln_kernel(const float* __restrict__ h,
                               const float* __restrict__ g,
                               const float* __restrict__ b,
                               unsigned short* __restrict__ outBf, int n) {
    int row = blockIdx.x * (blockDim.x >> 5) + (threadIdx.x >> 5);
    if (row >= n) return;
    int lane = threadIdx.x & 31;
    const float4 v4 = *(const float4*)(h + (long)row * 128 + lane * 4);
    float v[4] = {fmaxf(v4.x, 0.f), fmaxf(v4.y, 0.f), fmaxf(v4.z, 0.f), fmaxf(v4.w, 0.f)};
    float s = v[0] + v[1] + v[2] + v[3];
    float s2 = v[0]*v[0] + v[1]*v[1] + v[2]*v[2] + v[3]*v[3];
#pragma unroll
    for (int m = 1; m < 32; m <<= 1) {
        s  += __shfl_xor(s,  m, 32);
        s2 += __shfl_xor(s2, m, 32);
    }
    float mu  = s * (1.0f / 128.0f);
    float var = s2 * (1.0f / 128.0f) - mu * mu;
    float inv = rsqrtf(var + 1e-5f);
#pragma unroll
    for (int i = 0; i < 4; i++) {
        int c = lane * 4 + i;
        float y = (v[i] - mu) * inv * g[c] + b[c];
        outBf[(long)row * 128 + c] = f2bf(y);
    }
}

// ------------------------------- WMMA GEMM ---------------------------------
// D[n x 128] = A1 @ W1 (+ A2 @ W2) + bias, then optional L2-row-normalize /
// ReLU, then optional "outF += scale*D" (f32) and/or "outBf = bf16(D)".
// Block = 256 threads = 8 waves; wave w -> rows [blk*128+16w, +16) x 128 cols.
// HAS_A2 is a template parameter: the inner loop is branch-free so the
// scheduler can clause B-fragment loads across tiles and overlap with WMMA.

__device__ __forceinline__ v16bf load_a_frag(const unsigned short* rowp, int ks) {
    const v8bf lo = *(const v8bf*)(rowp + ks * 32);
    const v8bf hi = *(const v8bf*)(rowp + ks * 32 + 16);
    v16bf r;
#pragma unroll
    for (int i = 0; i < 8; i++) { r[i] = lo[i]; r[i + 8] = hi[i]; }
    return r;
}

template <bool HAS_A2>
__launch_bounds__(256)
__global__ void sage_gemm_kernel(const unsigned short* __restrict__ A1,
                                 const unsigned short* __restrict__ W1,
                                 const unsigned short* __restrict__ A2,
                                 const unsigned short* __restrict__ W2,
                                 const float* __restrict__ bias,
                                 float* __restrict__ outF,
                                 unsigned short* __restrict__ outBf,
                                 int n, float scale, int flags) {
    const int lane = threadIdx.x & 31;
    const int wave = threadIdx.x >> 5;
    const int rowBase = blockIdx.x * 128 + wave * 16;
    if (rowBase >= n) return;                 // wave-uniform: EXEC stays all-1s
    const int half = lane >> 4;
    const int l15  = lane & 15;
    long rowA = rowBase + l15;
    if (rowA >= n) rowA = n - 1;              // clamp tail loads
    const unsigned short* a1p = A1 + rowA * 128 + half * 8;
    const unsigned short* a2p = HAS_A2 ? (A2 + rowA * 128 + half * 8) : nullptr;

    v8f acc[8];
#pragma unroll
    for (int ct = 0; ct < 8; ct++) acc[ct] = (v8f){0.f,0.f,0.f,0.f,0.f,0.f,0.f,0.f};

#pragma unroll
    for (int ks = 0; ks < 4; ks++) {
        const v16bf a1 = load_a_frag(a1p, ks);
        v16bf a2;
        if (HAS_A2) a2 = load_a_frag(a2p, ks);
#pragma unroll
        for (int ct = 0; ct < 8; ct++) {
            const v16bf b1 = *(const v16bf*)(W1 + ((ct * 4 + ks) * 32 + lane) * 16);
            acc[ct] = __builtin_amdgcn_wmma_f32_16x16x32_bf16(
                false, a1, false, b1, (short)0, acc[ct], false, false);
            if (HAS_A2) {
                const v16bf b2 = *(const v16bf*)(W2 + ((ct * 4 + ks) * 32 + lane) * 16);
                acc[ct] = __builtin_amdgcn_wmma_f32_16x16x32_bf16(
                    false, a2, false, b2, (short)0, acc[ct], false, false);
            }
        }
    }

    // bias (per output column)
#pragma unroll
    for (int ct = 0; ct < 8; ct++) {
        float bv = bias[ct * 16 + l15];
#pragma unroll
        for (int r = 0; r < 8; r++) acc[ct][r] += bv;
    }

    if (flags & 2) {  // L2 row-normalize; a row lives in one VGPR across 16 lanes
        float ss[8];
#pragma unroll
        for (int r = 0; r < 8; r++) {
            float s = 0.f;
#pragma unroll
            for (int ct = 0; ct < 8; ct++) { float v = acc[ct][r]; s += v * v; }
            ss[r] = s;
        }
#pragma unroll
        for (int m = 1; m < 16; m <<= 1) {
#pragma unroll
            for (int r = 0; r < 8; r++) ss[r] += __shfl_xor(ss[r], m, 16);
        }
#pragma unroll
        for (int r = 0; r < 8; r++) {
            float inv = 1.0f / fmaxf(sqrtf(ss[r]), 1e-12f);
#pragma unroll
            for (int ct = 0; ct < 8; ct++) acc[ct][r] *= inv;
        }
    }
    if (flags & 1) {  // ReLU
#pragma unroll
        for (int ct = 0; ct < 8; ct++)
#pragma unroll
            for (int r = 0; r < 8; r++) acc[ct][r] = fmaxf(acc[ct][r], 0.f);
    }

    // C/D layout: VGPR r -> row r (lanes 0-15) / row 8+r (lanes 16-31), col = l15
#pragma unroll
    for (int r = 0; r < 8; r++) {
        int row = rowBase + half * 8 + r;
        if (row < n) {
            long base = (long)row * 128 + l15;
            if (outF) {
#pragma unroll
                for (int ct = 0; ct < 8; ct++) outF[base + ct * 16] += scale * acc[ct][r];
            }
            if (outBf) {
#pragma unroll
                for (int ct = 0; ct < 8; ct++) outBf[base + ct * 16] = f2bf(acc[ct][r]);
            }
        }
    }
}

// ------------------------------ orchestration ------------------------------

extern "C" void kernel_launch(void* const* d_in, const int* in_sizes, int n_in,
                              void* d_out, int out_size, void* d_ws, size_t ws_size,
                              hipStream_t stream) {
    (void)n_in; (void)out_size; (void)ws_size;

    const float* x    = (const float*)d_in[0];
    const int*   edges= (const int*)  d_in[1];
    const float* Wp   = (const float*)d_in[2];
    const float* bp   = (const float*)d_in[3];
    const float* Wl0  = (const float*)d_in[4];
    const float* bl0  = (const float*)d_in[5];
    const float* Wr0  = (const float*)d_in[6];
    const float* Wl   = (const float*)d_in[7];
    const float* bl   = (const float*)d_in[8];
    const float* Wr   = (const float*)d_in[9];
    const float* ln_g = (const float*)d_in[10];
    const float* ln_b = (const float*)d_in[11];

    const int  N  = in_sizes[0] / 128;
    const int  ET = in_sizes[2] / (128 * 128);
    const int  E  = (int)(in_sizes[1] / (ET * 2));
    const int  NR = in_sizes[7] / (ET * 128 * 128);   // L-1
    const long NC = (long)N * 128;
    const float invET = 1.0f / (float)ET;

    // workspace carve (256B-aligned bump allocator)
    char* p = (char*)d_ws;
    auto alloc = [&](size_t bytes) -> void* {
        void* r = (void*)p;
        p += (bytes + 255) & ~(size_t)255;
        return r;
    };
    unsigned short* xbf   = (unsigned short*)alloc(NC * 2);
    unsigned short* hbf   = (unsigned short*)alloc(NC * 2);
    unsigned short* hs    = (unsigned short*)alloc(NC * 2);
    unsigned short* aggbf = (unsigned short*)alloc(NC * 2);
    float*          agg   = (float*)alloc(NC * 4);
    float*          hacc  = (float*)alloc(NC * 4);
    float*          deg   = (float*)alloc((size_t)ET * N * 4);
    const int WM = 16384;  // elements per 128x128 matrix
    unsigned short* wsWp  = (unsigned short*)alloc((size_t)ET * WM * 2);
    unsigned short* wsWl0 = (unsigned short*)alloc((size_t)ET * WM * 2);
    unsigned short* wsWr0 = (unsigned short*)alloc((size_t)ET * WM * 2);
    unsigned short* wsWl  = (unsigned short*)alloc((size_t)NR * ET * WM * 2);
    unsigned short* wsWr  = (unsigned short*)alloc((size_t)NR * ET * WM * 2);

    const int TPB   = 256;
    const int gElem = (int)((NC + TPB - 1) / TPB);
    const int gGemm = (N + 127) / 128;
    const int gScat = (int)(((long)E * 32 + TPB - 1) / TPB);
    const int gEdge = (E + TPB - 1) / TPB;
    const int gRow  = (N + 7) / 8;
    const int gSwz  = (WM + TPB - 1) / TPB;
    const int gDegZ = (ET * N + TPB - 1) / TPB;

    // 1) swizzle all weights into bf16 B-fragment layout (L2-resident)
    for (int t = 0; t < ET; t++) {
        swz_kernel<<<gSwz, TPB, 0, stream>>>(Wp  + (long)t * WM, wsWp  + (long)t * WM);
        swz_kernel<<<gSwz, TPB, 0, stream>>>(Wl0 + (long)t * WM, wsWl0 + (long)t * WM);
        swz_kernel<<<gSwz, TPB, 0, stream>>>(Wr0 + (long)t * WM, wsWr0 + (long)t * WM);
    }
    for (int l = 0; l < NR; l++)
        for (int t = 0; t < ET; t++) {
            swz_kernel<<<gSwz, TPB, 0, stream>>>(Wl + (long)(l * ET + t) * WM,
                                                 wsWl + (long)(l * ET + t) * WM);
            swz_kernel<<<gSwz, TPB, 0, stream>>>(Wr + (long)(l * ET + t) * WM,
                                                 wsWr + (long)(l * ET + t) * WM);
        }

    // 2) x -> bf16; per-type degrees (reused across all layers)
    cvt_bf16_kernel<<<gElem, TPB, 0, stream>>>(x, xbf, NC);
    zero_kernel<<<gDegZ, TPB, 0, stream>>>(deg, (long)ET * N);
    for (int t = 0; t < ET; t++)
        deg_kernel<<<gEdge, TPB, 0, stream>>>(edges + ((long)t * 2 + 1) * E, deg + (long)t * N, E);

    // 3) layer 0: project=True, normalize=True
    zero_kernel<<<gElem, TPB, 0, stream>>>(hacc, NC);
    for (int t = 0; t < ET; t++) {
        // hs = relu(x @ Wp[t] + bp[t])
        sage_gemm_kernel<false><<<gGemm, TPB, 0, stream>>>(
            xbf, wsWp + (long)t * WM, nullptr, nullptr, bp + (long)t * 128,
            nullptr, hs, N, 1.0f, /*relu*/1);
        // agg = scatter-mean(hs[src] -> dst)
        zero_kernel<<<gElem, TPB, 0, stream>>>(agg, NC);
        scatter_kernel<<<gScat, TPB, 0, stream>>>(
            hs, edges + (long)t * 2 * E, edges + ((long)t * 2 + 1) * E, agg, E);
        finalize_kernel<<<gElem, TPB, 0, stream>>>(agg, deg + (long)t * N, aggbf, NC);
        // hacc += normalize(agg @ Wl0[t] + bl0[t] + x @ Wr0[t]) / ET
        sage_gemm_kernel<true><<<gGemm, TPB, 0, stream>>>(
            aggbf, wsWl0 + (long)t * WM, xbf, wsWr0 + (long)t * WM, bl0 + (long)t * 128,
            hacc, nullptr, N, invET, /*normalize*/2);
    }
    relu_ln_kernel<<<gRow, TPB, 0, stream>>>(hacc, ln_g, ln_b, hbf, N);

    // 4) remaining layers
    for (int l = 0; l < NR; l++) {
        const bool last = (l == NR - 1);
        float* target = last ? (float*)d_out : hacc;
        zero_kernel<<<gElem, TPB, 0, stream>>>(target, NC);
        for (int t = 0; t < ET; t++) {
            zero_kernel<<<gElem, TPB, 0, stream>>>(agg, NC);
            scatter_kernel<<<gScat, TPB, 0, stream>>>(
                hbf, edges + (long)t * 2 * E, edges + ((long)t * 2 + 1) * E, agg, E);
            finalize_kernel<<<gElem, TPB, 0, stream>>>(agg, deg + (long)t * N, aggbf, NC);
            sage_gemm_kernel<true><<<gGemm, TPB, 0, stream>>>(
                aggbf, wsWl + (long)(l * ET + t) * WM,
                hbf,   wsWr + (long)(l * ET + t) * WM,
                bl + (long)(l * ET + t) * 128,
                target, nullptr, N, invET, 0);
        }
        if (!last)
            relu_ln_kernel<<<gRow, TPB, 0, stream>>>(hacc, ln_g + (long)(l + 1) * 128,
                                                     ln_b + (long)(l + 1) * 128, hbf, N);
    }
}
